// WordAttention_11699490915081
// MI455X (gfx1250) — matrix operand order repaired and checked by reading, not compiled, not run on hardware
//
#include <hip/hip_runtime.h>

// ---------------------------------------------------------------------------
// WordAttention fused kernel for MI455X (gfx1250, wave32, WMMA f16->f32)
//   c:(32,2048,512) q:(32,512,512) W:(512,512) b:(512)  -> mix:(32,2048,512)
// Pre-pass builds f16 L2-resident operands (Wh, q_feat, qT); the main kernel
// fuses proj(ReLU) -> scores -> softmax -> mix per 32-row c tile, so c_feat /
// S / P never touch HBM (~320 MB HBM total).  B tiles are streamed into LDS
// with GLOBAL_LOAD_ASYNC_TO_LDS_B128 (ASYNCcnt), double-buffered so the L2
// stream overlaps the v_wmma issue.
// ---------------------------------------------------------------------------

#define DIM   512
#define BATCH 32
#define CLEN  2048
#define QLEN  512

typedef _Float16 h4  __attribute__((ext_vector_type(4)));
typedef _Float16 h8  __attribute__((ext_vector_type(8)));
typedef _Float16 h16 __attribute__((ext_vector_type(16)));
typedef float    f8  __attribute__((ext_vector_type(8)));

#define AF_STRIDE 520   // halfs: 512 + 8 pad  (1040 B row -> conflict-free b128)
#define BS_STRIDE 40    // halfs: 32 + 8 pad   (80 B row  -> conflict-free b128)

// LDS byte offset of a __shared__ object (addrspace(3) ptrtoint).
typedef __attribute__((address_space(3))) void lds_void;
#define LDSOFF(p) ((uint32_t)(uintptr_t)(lds_void*)(void*)(p))

__device__ __forceinline__ f8 f8zero() {
  f8 z;
#pragma unroll
  for (int i = 0; i < 8; ++i) z[i] = 0.0f;
  return z;
}

__device__ __forceinline__ h16 cat8(h8 lo, h8 hi) {
  return __builtin_shufflevector(lo, hi, 0,1,2,3,4,5,6,7,8,9,10,11,12,13,14,15);
}

// A fragment (16x32 f16, MxK): lanes 0-15 row M=lane hold K {0..7,16..23},
// lanes 16-31 hold K {8..15,24..31}.
__device__ __forceinline__ h16 load_a_frag(const _Float16* base, int lane) {
  const int m  = lane & 15;
  const int k0 = (lane < 16) ? 0 : 8;
  const _Float16* p = base + m * AF_STRIDE + k0;
  h8 lo = *(const h8*)(p);
  h8 hi = *(const h8*)(p + 16);
  return cat8(lo, hi);
}

// B fragment (32x16 f16, KxN) from Bs stored [n][k]: lane n<16 holds K
// {0..7,16..23} of column n, lane n+16 holds K {8..15,24..31}.
__device__ __forceinline__ h16 load_b_frag(const _Float16* base, int lane) {
  const int n  = lane & 15;
  const int k0 = (lane < 16) ? 0 : 8;
  const _Float16* p = base + n * BS_STRIDE + k0;
  h8 lo = *(const h8*)(p);
  h8 hi = *(const h8*)(p + 16);
  return cat8(lo, hi);
}

__device__ __forceinline__ f8 wmma16(h16 a, h16 b, f8 c) {
  return __builtin_amdgcn_wmma_f32_16x16x32_f16(false, a, false, b,
                                                (short)0, c, false, false);
}

// Async-copy one 64 B row (4x b128); ISA adds the literal offset to BOTH the
// LDS and the global address, so row-contiguous chunks map 1:1.
__device__ __forceinline__ void async_row64(uint32_t lds, uint32_t goff,
                                            const void* sbase) {
  asm volatile(
      "global_load_async_to_lds_b128 %0, %1, %2\n\t"
      "global_load_async_to_lds_b128 %0, %1, %2 offset:16\n\t"
      "global_load_async_to_lds_b128 %0, %1, %2 offset:32\n\t"
      "global_load_async_to_lds_b128 %0, %1, %2 offset:48"
      :: "v"(lds), "v"(goff), "s"(sbase) : "memory");
}

__device__ __forceinline__ void wait_async0() {
  asm volatile("s_wait_asynccnt 0" ::: "memory");
}

// Stage one 512-row x 32-half B tile (32 KB): thread t copies rows t, t+256.
__device__ __forceinline__ void stage_async(const _Float16* Bg,
                                            uint32_t bs_lds, int k0, int tid) {
  const uint32_t go0 = (uint32_t)tid * (DIM * 2) + (uint32_t)k0 * 2;
  const uint32_t lo0 = bs_lds + (uint32_t)tid * (BS_STRIDE * 2);
  async_row64(lo0, go0, Bg);
  async_row64(lo0 + 256u * (BS_STRIDE * 2), go0 + 256u * (DIM * 2), Bg);
}

// 32(M) x 512(N) x 512(K) GEMM.  A: f16 tile in LDS (stride AF_STRIDE).
// B: global f16 row-major [512][512] read as [n][k], async-streamed into a
// double-buffered LDS tile one K-step ahead.  8 waves = 2(M) x 4(N); each
// wave owns 16 rows x 128 cols = 8 accumulator fragments.
__device__ __forceinline__ void gemm_32x512(
    const _Float16* A_lds, const _Float16* __restrict__ Bg,
    _Float16* Bs, f8 acc[8])
{
  const int tid  = threadIdx.x;
  const int lane = tid & 31;
  const int wave = tid >> 5;
  const int wm   = wave >> 2;   // 0..1
  const int wn   = wave & 3;    // 0..3
  const uint32_t bs0 = LDSOFF(Bs);
#pragma unroll
  for (int t = 0; t < 8; ++t) acc[t] = f8zero();

  // prologue: stage K-step 0 into buffer 0
  stage_async(Bg, bs0, 0, tid);
  wait_async0();
  __syncthreads();

  for (int it = 0; it < DIM / 32; ++it) {
    const int k0 = it * 32;
    if (it + 1 < DIM / 32)   // prefetch next K-step into the other buffer
      stage_async(Bg, bs0 + (uint32_t)((it + 1) & 1) * (512 * BS_STRIDE * 2),
                  k0 + 32, tid);
    const _Float16* Bcur = Bs + (it & 1) * (512 * BS_STRIDE);
    h16 a = load_a_frag(A_lds + wm * 16 * AF_STRIDE + k0, lane);
    const _Float16* bb = Bcur + (wn * 128) * BS_STRIDE;
#pragma unroll
    for (int t = 0; t < 8; ++t) {
      h16 b = load_b_frag(bb + t * 16 * BS_STRIDE, lane);
      acc[t] = wmma16(a, b, acc[t]);
    }
    wait_async0();     // drain prefetch issued this iteration
    __syncthreads();   // make all waves' staged data visible / guard reuse
  }
}

// ------------------------------- pre-pass ----------------------------------

__global__ void cvt_w_kernel(const float* __restrict__ W,
                             _Float16* __restrict__ Wh) {
  const int i = (blockIdx.x * 256 + threadIdx.x) * 4;
  const float4 v = *(const float4*)(W + i);
  h4 o; o[0] = (_Float16)v.x; o[1] = (_Float16)v.y;
        o[2] = (_Float16)v.z; o[3] = (_Float16)v.w;
  *(h4*)(Wh + i) = o;
}

__global__ __launch_bounds__(256) void tq_kernel(const float* __restrict__ q,
                                                 _Float16* __restrict__ qT) {
  __shared__ _Float16 t[32][33];
  const int b = blockIdx.z, i0 = blockIdx.y * 32, j0 = blockIdx.x * 32;
  const int tid = threadIdx.x;
  for (int idx = tid; idx < 1024; idx += 256) {
    const int i = idx >> 5, j = idx & 31;
    t[i][j] = (_Float16)q[(((size_t)b * QLEN) + i0 + i) * DIM + j0 + j];
  }
  __syncthreads();
  for (int idx = tid; idx < 1024; idx += 256) {
    const int j = idx >> 5, i = idx & 31;
    qT[(((size_t)b * DIM) + j0 + j) * QLEN + i0 + i] = t[i][j];
  }
}

// q_feat = relu(q @ W^T + bias), f16 row-major
__global__ __launch_bounds__(256) void qproj_kernel(
    const float* __restrict__ q, const float* __restrict__ bias,
    const _Float16* __restrict__ Wh, _Float16* __restrict__ qfeat)
{
  __shared__ alignas(16) _Float16 Af[32 * AF_STRIDE];
  __shared__ alignas(16) _Float16 Bs[2 * 512 * BS_STRIDE];
  const int tid = threadIdx.x, lane = tid & 31, wave = tid >> 5;
  const int wm = wave >> 2, wn = wave & 3;
  const int b = blockIdx.x >> 4;               // 16 row-blocks per batch
  const int row0 = (blockIdx.x & 15) * 32;

  const float* qbase = q + ((size_t)b * QLEN + row0) * DIM;
  for (int idx = tid; idx < 32 * 128; idx += 256) {
    const int r = idx >> 7, c4 = (idx & 127) << 2;
    const float4 v = *(const float4*)(qbase + (size_t)r * DIM + c4);
    _Float16* d = Af + r * AF_STRIDE + c4;
    d[0]=(_Float16)v.x; d[1]=(_Float16)v.y; d[2]=(_Float16)v.z; d[3]=(_Float16)v.w;
  }
  __syncthreads();

  f8 acc[8];
  gemm_32x512(Af, Wh, Bs, acc);

  const int rbase = wm * 16 + ((lane < 16) ? 0 : 8);
  const int cb = wn * 128 + (lane & 15);
  _Float16* obase = qfeat + ((size_t)b * QLEN + row0) * DIM;
#pragma unroll
  for (int t = 0; t < 8; ++t) {
    const float bv = bias[cb + t * 16];
#pragma unroll
    for (int v = 0; v < 8; ++v) {
      float x = acc[t][v] + bv;
      x = x > 0.0f ? x : 0.0f;
      obase[(size_t)(rbase + v) * DIM + cb + t * 16] = (_Float16)x;
    }
  }
}

// ------------------------------ main fused ---------------------------------

struct __align__(16) SharedMem {
  _Float16 Af[32 * AF_STRIDE];       // c tile (GEMM1 A), reused as P (GEMM3 A)
  _Float16 Cf[32 * AF_STRIDE];       // c_feat tile (GEMM2 A)
  _Float16 Bs[2 * 512 * BS_STRIDE];  // double-buffered streamed B tile
  float red[32][4];
  float rowstat[32];
};

__global__ __launch_bounds__(256) void fused_attn_kernel(
    const float* __restrict__ c, const float* __restrict__ bias,
    const _Float16* __restrict__ Wh, const _Float16* __restrict__ qfeat,
    const _Float16* __restrict__ qT, float* __restrict__ out)
{
  __shared__ SharedMem sm;
  const int tid = threadIdx.x, lane = tid & 31, wave = tid >> 5;
  const int wm = wave >> 2, wn = wave & 3;
  const int b = blockIdx.x >> 6;               // 64 row-blocks per batch
  const int row0 = (blockIdx.x & 63) * 32;
  const int rbase = wm * 16 + ((lane < 16) ? 0 : 8);
  const int cb = wn * 128 + (lane & 15);

  // stage 32x512 c tile -> f16 LDS
  const float* cbase = c + ((size_t)b * CLEN + row0) * DIM;
  for (int idx = tid; idx < 32 * 128; idx += 256) {
    const int r = idx >> 7, c4 = (idx & 127) << 2;
    const float4 v = *(const float4*)(cbase + (size_t)r * DIM + c4);
    _Float16* d = sm.Af + r * AF_STRIDE + c4;
    d[0]=(_Float16)v.x; d[1]=(_Float16)v.y; d[2]=(_Float16)v.z; d[3]=(_Float16)v.w;
  }
  __syncthreads();

  f8 acc[8];

  // GEMM1: c_feat = relu(c @ W^T + bias) -> Cf (f16)
  gemm_32x512(sm.Af, Wh, sm.Bs, acc);
#pragma unroll
  for (int t = 0; t < 8; ++t) {
    const float bv = bias[cb + t * 16];
#pragma unroll
    for (int v = 0; v < 8; ++v) {
      float x = acc[t][v] + bv;
      x = x > 0.0f ? x : 0.0f;
      sm.Cf[(rbase + v) * AF_STRIDE + cb + t * 16] = (_Float16)x;
    }
  }
  __syncthreads();

  // GEMM2: S = c_feat @ q_feat^T  (f32 accum, 16x128 per wave in registers)
  gemm_32x512(sm.Cf, qfeat + (size_t)b * QLEN * DIM, sm.Bs, acc);

  // row softmax over 512 cols: shuffle-reduce within 16-lane halves, then
  // cross-wave combine in LDS.  P = exp(S - max) unnormalized -> Af; 1/rowsum
  // applied in GEMM3 epilogue (linearity of P @ q).
#pragma unroll
  for (int v = 0; v < 8; ++v) {
    float m = acc[0][v];
#pragma unroll
    for (int t = 1; t < 8; ++t) m = fmaxf(m, acc[t][v]);
    for (int s = 8; s >= 1; s >>= 1) m = fmaxf(m, __shfl_xor(m, s, 32));
    if ((lane & 15) == 0) sm.red[rbase + v][wn] = m;
  }
  __syncthreads();
  if (tid < 32)
    sm.rowstat[tid] = fmaxf(fmaxf(sm.red[tid][0], sm.red[tid][1]),
                            fmaxf(sm.red[tid][2], sm.red[tid][3]));
  __syncthreads();
#pragma unroll
  for (int v = 0; v < 8; ++v) {
    const float rmax = sm.rowstat[rbase + v];
    float s = 0.0f;
#pragma unroll
    for (int t = 0; t < 8; ++t) {
      const float e = __expf(acc[t][v] - rmax);
      sm.Af[(rbase + v) * AF_STRIDE + cb + t * 16] = (_Float16)e;
      s += e;
    }
    for (int sh = 8; sh >= 1; sh >>= 1) s += __shfl_xor(s, sh, 32);
    if ((lane & 15) == 0) sm.red[rbase + v][wn] = s;
  }
  __syncthreads();
  if (tid < 32)
    sm.rowstat[tid] = 1.0f / (sm.red[tid][0] + sm.red[tid][1] +
                              sm.red[tid][2] + sm.red[tid][3]);
  __syncthreads();

  // GEMM3: mix = (P @ q) * (1/rowsum), via qT so B is row-major [d][q_len]
  gemm_32x512(sm.Af, qT + (size_t)b * DIM * QLEN, sm.Bs, acc);
  float* obase = out + ((size_t)b * CLEN + row0) * DIM;
#pragma unroll
  for (int v = 0; v < 8; ++v) {
    const float inv = sm.rowstat[rbase + v];
#pragma unroll
    for (int t = 0; t < 8; ++t)
      obase[(size_t)(rbase + v) * DIM + cb + t * 16] = acc[t][v] * inv;
  }
}

// ------------------------------- launcher ----------------------------------

extern "C" void kernel_launch(void* const* d_in, const int* in_sizes, int n_in,
                              void* d_out, int out_size, void* d_ws, size_t ws_size,
                              hipStream_t stream) {
  (void)in_sizes; (void)n_in; (void)out_size; (void)ws_size;
  const float* c    = (const float*)d_in[0];
  const float* q    = (const float*)d_in[1];
  const float* W    = (const float*)d_in[2];
  const float* bias = (const float*)d_in[3];
  float* out = (float*)d_out;

  _Float16* Wh    = (_Float16*)d_ws;                       // 512*512 halfs
  _Float16* qfeat = Wh + (size_t)DIM * DIM;                // 32*512*512 halfs
  _Float16* qT    = qfeat + (size_t)BATCH * QLEN * DIM;    // 32*512*512 halfs

  cvt_w_kernel<<<DIM * DIM / (256 * 4), 256, 0, stream>>>(W, Wh);
  tq_kernel<<<dim3(DIM / 32, QLEN / 32, BATCH), 256, 0, stream>>>(q, qT);
  qproj_kernel<<<BATCH * (QLEN / 32), 256, 0, stream>>>(q, bias, Wh, qfeat);
  fused_attn_kernel<<<BATCH * (CLEN / 32), 256, 0, stream>>>(c, bias, Wh,
                                                             qfeat, qT, out);
}
